// MaskDINODecoder_19258633355694
// MI455X (gfx1250) — matrix operand order repaired and checked by reading, not compiled, not run on hardware
//
#include <hip/hip_runtime.h>
#include <hip/hip_bf16.h>
#include <math.h>

// ---------------------------------------------------------------------------
// MaskDINO decoder for MI455X (gfx1250, wave32, WMMA).
// Every matmul lowers to v_wmma_f32_16x16x32_bf16. One wave computes a 16x32
// output strip (2 WMMA per k-step sharing one A fragment). Weights are
// transposed to bf16 [Npad32][Kpad32] so each B fragment is two 16B
// loads per lane. The dominant GEMM (value projection, 43520x256x256,
// ~70% of FLOPs) reads a once-converted bf16 copy of `memory` (L2-resident).
// ---------------------------------------------------------------------------

typedef __attribute__((ext_vector_type(16))) __bf16 v16bf;
typedef __attribute__((ext_vector_type(8)))  __bf16 v8bf;
typedef __attribute__((ext_vector_type(8)))  float  v8f;

constexpr int Bc  = 2;
constexpr int NQ  = 300;
constexpr int Cc  = 256;
constexpr int Hh  = 8;
constexpr int DH  = 32;      // head dim == wave32 lanes
constexpr int LVLn= 4;
constexpr int Pp  = 4;
constexpr int DFF = 2048;
constexpr int NLn = 9;
constexpr int LM  = 21760;   // 128^2+64^2+32^2+16^2
constexpr int BQ  = Bc*NQ;   // 600
constexpr int BHn = Bc*Hh;   // 16
constexpr int KEYP32 = 320;  // keys padded to mult of 32 (N of QK^T, K of P*V)

__device__ __forceinline__ __bf16 f2bf(float x) {
  unsigned u = __builtin_bit_cast(unsigned, x);
  unsigned r = u + 0x7FFFu + ((u >> 16) & 1u);       // round-to-nearest-even
  unsigned short s = (unsigned short)(r >> 16);
  return __builtin_bit_cast(__bf16, s);
}

__device__ __forceinline__ float waveSum(float v) {
  #pragma unroll
  for (int o = 16; o > 0; o >>= 1) v += __shfl_xor(v, o, 32);
  return v;
}
__device__ __forceinline__ float waveMax(float v) {
  #pragma unroll
  for (int o = 16; o > 0; o >>= 1) v = fmaxf(v, __shfl_xor(v, o, 32));
  return v;
}

// A-fragment loaders (16x32 bf16 tile striped per ISA layout):
// lane m = lane&15, half = lane>>4:
//   elems 0..7  -> K = kk + 8*half + i        (contiguous)
//   elems 8..15 -> K = kk + 16 + 8*half + i   (contiguous)
__device__ __forceinline__ v16bf loadAfrag(const float* __restrict__ Arow,
                                           int kk, int half) {
  const float4* p1 = reinterpret_cast<const float4*>(Arow + kk + 8*half);
  const float4* p2 = reinterpret_cast<const float4*>(Arow + kk + 16 + 8*half);
  float4 r0 = p1[0], r1 = p1[1], r2 = p2[0], r3 = p2[1];
  v16bf a;
  a[0]=f2bf(r0.x); a[1]=f2bf(r0.y); a[2]=f2bf(r0.z); a[3]=f2bf(r0.w);
  a[4]=f2bf(r1.x); a[5]=f2bf(r1.y); a[6]=f2bf(r1.z); a[7]=f2bf(r1.w);
  a[8]=f2bf(r2.x); a[9]=f2bf(r2.y); a[10]=f2bf(r2.z); a[11]=f2bf(r2.w);
  a[12]=f2bf(r3.x); a[13]=f2bf(r3.y); a[14]=f2bf(r3.z); a[15]=f2bf(r3.w);
  return a;
}
__device__ __forceinline__ v16bf loadAfrag(const __bf16* __restrict__ Arow,
                                           int kk, int half) {
  v8bf r0 = *reinterpret_cast<const v8bf*>(Arow + kk + 8*half);
  v8bf r1 = *reinterpret_cast<const v8bf*>(Arow + kk + 16 + 8*half);
  v16bf a;
  #pragma unroll
  for (int i = 0; i < 8; ++i) { a[i] = r0[i]; a[8+i] = r1[i]; }
  return a;
}

// ---------------------------------------------------------------------------
// Batched GEMM:  C[b] = act( alpha * A[b](MxK) @ W(KxN) + bias )
// A row-major (fp32 or bf16), K padded to 32 & zero-filled by producers.
// WT bf16 = W transposed, [Npad32][Kpad32], zero-filled pads.
// One wave -> 16x32 output strip; blockDim=(32,4); grid=(M/64, N/32, batch).
// Out-of-range M rows are row-clamped on load (their D rows are not stored),
// keeping the inner loop branch-free with EXEC all-ones for WMMA.
// ---------------------------------------------------------------------------
template<bool RELU, typename AT>
__global__ void gemm_wmma(const AT* __restrict__ A,
                          const __bf16* __restrict__ WT,
                          const float* __restrict__ bias,
                          float* __restrict__ Cout,
                          int M, int N, int Kp, int lda, int ldwt, int ldc,
                          long long strA, long long strW, long long strC,
                          float alpha) {
  const int batch = blockIdx.z;
  const AT*     Ab = A    + (long long)batch * strA;
  const __bf16* Wb = WT   + (long long)batch * strW;
  float*        Cb = Cout + (long long)batch * strC;

  const int mTile = blockIdx.x * blockDim.y + threadIdx.y;
  if (mTile * 16 >= M) return;                       // uniform per wave
  const int lane  = threadIdx.x;
  const int half  = lane >> 4;
  const int mrow  = min(mTile * 16 + (lane & 15), M - 1);  // branchless clamp
  const int ncol0 = blockIdx.y * 32 + (lane & 15);

  const AT*     Arow = Ab + (long long)mrow * lda;
  const __bf16* Wr0  = Wb + (long long)ncol0 * ldwt;
  const __bf16* Wr1  = Wr0 + (long long)16 * ldwt;

  v8f acc[2];
  acc[0] = (v8f){0.f,0.f,0.f,0.f,0.f,0.f,0.f,0.f};
  acc[1] = (v8f){0.f,0.f,0.f,0.f,0.f,0.f,0.f,0.f};

  for (int kk = 0; kk < Kp; kk += 32) {
    v16bf a  = loadAfrag(Arow, kk, half);
    v16bf b0 = __builtin_shufflevector(*reinterpret_cast<const v8bf*>(Wr0 + kk + 8*half), *reinterpret_cast<const v8bf*>(Wr0 + kk + 16 + 8*half), 0,1,2,3,4,5,6,7,8,9,10,11,12,13,14,15);
    v16bf b1 = __builtin_shufflevector(*reinterpret_cast<const v8bf*>(Wr1 + kk + 8*half), *reinterpret_cast<const v8bf*>(Wr1 + kk + 16 + 8*half), 0,1,2,3,4,5,6,7,8,9,10,11,12,13,14,15);
    __builtin_prefetch(Arow + kk + 32, 0, 0);        // speculative, dropped if OOB
    acc[0] = __builtin_amdgcn_wmma_f32_16x16x32_bf16(false, a, false, b0,
                                                     (short)0, acc[0], false, false);
    acc[1] = __builtin_amdgcn_wmma_f32_16x16x32_bf16(false, a, false, b1,
                                                     (short)0, acc[1], false, false);
  }

  #pragma unroll
  for (int t = 0; t < 2; ++t) {
    int nc = ncol0 + 16*t;
    if (nc < N) {
      const float bv = bias ? bias[nc] : 0.f;
      #pragma unroll
      for (int v = 0; v < 8; ++v) {
        int r = mTile * 16 + v + 8*half;
        if (r < M) {
          float xv = acc[t][v] * alpha + bv;
          if (RELU) xv = fmaxf(xv, 0.f);
          Cb[(long long)r * ldc + nc] = xv;
        }
      }
    }
  }
}

// W fp32 [K][N] -> WT bf16 [Np][Kp], zero padded (Np mult 32, Kp mult 32).
__global__ void convert_wt(const float* __restrict__ W, __bf16* __restrict__ WT,
                           int K, int N, int Kp, int Np) {
  int idx = blockIdx.x * blockDim.x + threadIdx.x;
  if (idx >= Np * Kp) return;
  int n = idx / Kp, k = idx % Kp;
  float v = (n < N && k < K) ? W[(long long)k * N + n] : 0.f;
  WT[idx] = f2bf(v);
}

// row-major fp32 -> bf16 (for the once-per-call memory conversion)
__global__ void convert_bf(const float* __restrict__ in, __bf16* __restrict__ out,
                           int n) {
  int i = blockIdx.x * blockDim.x + threadIdx.x;
  if (i < n) out[i] = f2bf(in[i]);
}

__global__ void ew_add(const float* a, const float* b, float* o, int n) {
  int i = blockIdx.x * blockDim.x + threadIdx.x;
  if (i < n) o[i] = a[i] + b[i];
}

__global__ void sigmoid0(const float* x, float* refcur, float* refs0, int n) {
  int i = blockIdx.x * blockDim.x + threadIdx.x;
  if (i < n) { float s = 1.f / (1.f + expf(-x[i])); refcur[i] = s; refs0[i] = s; }
}

// sine position embedding of ref_in[:, :, 0, :]  -> qse [BQ][512]
__global__ void sine_embed_k(const float* __restrict__ ref,
                             const float* __restrict__ vratio,
                             float* __restrict__ qse) {
  int bq = blockIdx.x;                  // BQ
  int j  = threadIdx.x;                 // 0..127
  int b  = bq / NQ;
  float vr0 = vratio[(b*LVLn + 0)*2 + 0];
  float vr1 = vratio[(b*LVLn + 0)*2 + 1];
  const float* r = ref + (long long)bq * 4;
  float comp[4] = { r[1]*vr1, r[0]*vr0, r[2]*vr0, r[3]*vr1 };  // y, x, w, h
  float freq = powf(10000.f, (float)(j >> 1) / 64.f);
  float sc = 6.283185307179586f / freq;
  #pragma unroll
  for (int c = 0; c < 4; ++c) {
    float t = comp[c] * sc;
    qse[(long long)bq * 512 + c*128 + j] = (j & 1) ? cosf(t) : sinf(t);
  }
}

// split qkv into WMMA-ready attention operands
__global__ void prep_qh(const float* __restrict__ qk, float* __restrict__ qh) {
  int idx = blockIdx.x * blockDim.x + threadIdx.x;
  if (idx >= BHn*NQ*DH) return;
  int d = idx & 31, q = (idx >> 5) % NQ, bh = idx / (NQ*DH);
  int b = bh >> 3, h = bh & 7;
  qh[idx] = qk[((long long)(b*NQ + q))*512 + h*DH + d];
}
__global__ void prep_kh(const float* __restrict__ qk, __bf16* __restrict__ khT) {
  int idx = blockIdx.x * blockDim.x + threadIdx.x;
  if (idx >= BHn*KEYP32*DH) return;
  int d = idx & 31, key = (idx >> 5) % KEYP32, bh = idx / (KEYP32*DH);
  int b = bh >> 3, h = bh & 7;
  float v = (key < NQ) ? qk[((long long)(b*NQ + key))*512 + 256 + h*DH + d] : 0.f;
  khT[idx] = f2bf(v);
}
__global__ void prep_vh(const float* __restrict__ vb, __bf16* __restrict__ vhT) {
  int idx = blockIdx.x * blockDim.x + threadIdx.x;
  if (idx >= BHn*DH*KEYP32) return;
  int key = idx % KEYP32, d = (idx / KEYP32) & 31, bh = idx / (KEYP32*DH);
  int b = bh >> 3, h = bh & 7;
  float v = (key < NQ) ? vb[((long long)(b*NQ + key))*Cc + h*DH + d] : 0.f;
  vhT[idx] = f2bf(v);
}
__global__ void merge_heads(const float* __restrict__ ao, float* __restrict__ o) {
  int idx = blockIdx.x * blockDim.x + threadIdx.x;
  if (idx >= BHn*NQ*DH) return;
  int d = idx & 31, q = (idx >> 5) % NQ, bh = idx / (NQ*DH);
  int b = bh >> 3, h = bh & 7;
  o[((long long)(b*NQ + q))*Cc + h*DH + d] = ao[idx];
}

// softmax over 300 keys (row stride KEYP32, pads zeroed) — 10 waves/row
__global__ void attn_softmax(float* __restrict__ sc) {
  __shared__ float sm[10];
  int row = blockIdx.x, t = threadIdx.x;
  float* p = sc + (long long)row * KEYP32;
  float v = (t < NQ) ? p[t] : -1e30f;
  float m = waveMax(v);
  int w = t >> 5, lane = t & 31;
  if (lane == 0) sm[w] = m;
  __syncthreads();
  if (t == 0) { float mm = sm[0]; for (int i = 1; i < 10; ++i) mm = fmaxf(mm, sm[i]); sm[0] = mm; }
  __syncthreads();
  m = sm[0];
  __syncthreads();
  float e = (t < NQ) ? expf(v - m) : 0.f;
  float s = waveSum(e);
  if (lane == 0) sm[w] = s;
  __syncthreads();
  if (t == 0) { float ss = 0.f; for (int i = 0; i < 10; ++i) ss += sm[i]; sm[0] = ss; }
  __syncthreads();
  p[t] = e / sm[0];
}

// softmax over 16 (LVL*P) per (b,q,h)
__global__ void aw_softmax(float* __restrict__ aw) {
  int i = blockIdx.x * blockDim.x + threadIdx.x;
  if (i >= BQ * Hh) return;
  float* p = aw + (long long)(i / Hh) * 128 + (i % Hh) * 16;
  float m = -1e30f;
  #pragma unroll
  for (int j = 0; j < 16; ++j) m = fmaxf(m, p[j]);
  float s = 0.f; float e[16];
  #pragma unroll
  for (int j = 0; j < 16; ++j) { e[j] = expf(p[j] - m); s += e[j]; }
  float inv = 1.f / s;
  #pragma unroll
  for (int j = 0; j < 16; ++j) p[j] = e[j] * inv;
}

// out = LayerNorm(x (+ y)) * g + b ; C=256, blockDim=256
__global__ void add_ln(const float* __restrict__ x, const float* __restrict__ y,
                       const float* __restrict__ g, const float* __restrict__ bt,
                       float* __restrict__ out) {
  __shared__ float s1[8], s2[8];
  int row = blockIdx.x, t = threadIdx.x;
  long long base = (long long)row * Cc + t;
  float v = x[base] + (y ? y[base] : 0.f);
  float a = waveSum(v), b2 = waveSum(v * v);
  int w = t >> 5, lane = t & 31;
  if (lane == 0) { s1[w] = a; s2[w] = b2; }
  __syncthreads();
  if (t == 0) {
    float aa = 0.f, bb = 0.f;
    for (int i = 0; i < 8; ++i) { aa += s1[i]; bb += s2[i]; }
    s1[0] = aa; s2[0] = bb;
  }
  __syncthreads();
  float mean = s1[0] / 256.f;
  float var  = s2[0] / 256.f - mean * mean;
  float inv  = rsqrtf(var + 1e-5f);
  out[base] = (v - mean) * inv * g[t] + bt[t];
}

// multi-scale deformable sampling: one wave per (b,q,h), lane = channel d
__global__ void ms_deform(const float* __restrict__ value,
                          const float* __restrict__ so,
                          const float* __restrict__ awsm,
                          const float* __restrict__ ref,
                          const float* __restrict__ vratio,
                          float* __restrict__ out) {
  const int bq = blockIdx.x, b = bq / NQ;
  const int d = threadIdx.x, h = threadIdx.y;
  const int HHa[4] = {128, 64, 32, 16};
  const int STa[4] = {0, 16384, 20480, 21504};
  const float* r = ref + (long long)bq * 4;
  float accum = 0.f;
  #pragma unroll
  for (int l = 0; l < 4; ++l) {
    float vr0 = vratio[(b*LVLn + l)*2 + 0];
    float vr1 = vratio[(b*LVLn + l)*2 + 1];
    float cx = r[0]*vr0, cy = r[1]*vr1, cw = r[2]*vr0, ch = r[3]*vr1;
    int hh = HHa[l], ww = HHa[l];
    const float* vbase = value + ((long long)b*LM + STa[l]) * Cc + h*DH + d;
    #pragma unroll
    for (int p = 0; p < 4; ++p) {
      long long oidx = (long long)bq * 256 + (((h*LVLn + l)*Pp + p) * 2);
      float gx = cx + so[oidx + 0] * 0.25f * cw * 0.5f;
      float gy = cy + so[oidx + 1] * 0.25f * ch * 0.5f;
      float x = gx * ww - 0.5f, y = gy * hh - 0.5f;
      float x0f = floorf(x), y0f = floorf(y);
      int x0 = (int)x0f, y0 = (int)y0f;
      float wx = x - x0f, wy = y - y0f;
      float s = 0.f;
      #pragma unroll
      for (int cyy = 0; cyy < 2; ++cyy) {
        #pragma unroll
        for (int cxx = 0; cxx < 2; ++cxx) {
          int xx = x0 + cxx, yy = y0 + cyy;
          if (xx >= 0 && xx < ww && yy >= 0 && yy < hh) {
            float wgt = (cxx ? wx : 1.f - wx) * (cyy ? wy : 1.f - wy);
            s += wgt * vbase[((long long)yy * ww + xx) * Cc];
          }
        }
      }
      accum += s * awsm[(long long)bq * 128 + h*16 + l*4 + p];
    }
  }
  out[(long long)bq * Cc + h*DH + d] = accum;
}

// ref update: sig = sigmoid(delta + inverse_sigmoid(ref)); write refs[i+1]
__global__ void update_ref(const float* __restrict__ delta,
                           float* __restrict__ refcur,
                           float* __restrict__ refs_out) {
  int i = blockIdx.x * blockDim.x + threadIdx.x;
  if (i >= BQ * 4) return;
  float r = fminf(fmaxf(refcur[i], 0.f), 1.f);
  float inv = logf(fmaxf(r, 1e-3f) / fmaxf(1.f - r, 1e-3f));
  float s = 1.f / (1.f + expf(-(delta[i] + inv)));
  refcur[i] = s;
  refs_out[i] = s;
}

// ---------------------------------------------------------------------------
extern "C" void kernel_launch(void* const* d_in, const int* in_sizes, int n_in,
                              void* d_out, int out_size, void* d_ws, size_t ws_size,
                              hipStream_t stream) {
  (void)in_sizes; (void)n_in; (void)out_size; (void)ws_size;

  // inputs (setup_inputs insertion order, params flattened in insertion order)
  const float* tgt_in  = (const float*)d_in[0];
  const float* memory  = (const float*)d_in[1];
  const float* rpu     = (const float*)d_in[2];
  const float* vratio  = (const float*)d_in[3];
  int pi = 4;
  const float* so_w  = (const float*)d_in[pi++]; const float* so_b  = (const float*)d_in[pi++];
  const float* aw_w  = (const float*)d_in[pi++]; const float* aw_b  = (const float*)d_in[pi++];
  const float* vp_w  = (const float*)d_in[pi++]; const float* vp_b  = (const float*)d_in[pi++];
  const float* op_w  = (const float*)d_in[pi++]; const float* op_b  = (const float*)d_in[pi++];
  const float* sain_w= (const float*)d_in[pi++]; const float* sain_b= (const float*)d_in[pi++];
  const float* saout_w=(const float*)d_in[pi++]; const float* saout_b=(const float*)d_in[pi++];
  const float* n1_g  = (const float*)d_in[pi++]; const float* n1_b  = (const float*)d_in[pi++];
  const float* n2_g  = (const float*)d_in[pi++]; const float* n2_b  = (const float*)d_in[pi++];
  const float* n3_g  = (const float*)d_in[pi++]; const float* n3_b  = (const float*)d_in[pi++];
  const float* ff1_w = (const float*)d_in[pi++]; const float* ff1_b = (const float*)d_in[pi++];
  const float* ff2_w = (const float*)d_in[pi++]; const float* ff2_b = (const float*)d_in[pi++];
  const float* bb_w1 = (const float*)d_in[pi++]; const float* bb_b1 = (const float*)d_in[pi++];
  const float* bb_w2 = (const float*)d_in[pi++]; const float* bb_b2 = (const float*)d_in[pi++];
  const float* bb_w3 = (const float*)d_in[pi++]; const float* bb_b3 = (const float*)d_in[pi++];
  const float* rph_w1= (const float*)d_in[pi++]; const float* rph_b1= (const float*)d_in[pi++];
  const float* rph_w2= (const float*)d_in[pi++]; const float* rph_b2= (const float*)d_in[pi++];
  const float* dn_g  = (const float*)d_in[pi++]; const float* dn_b  = (const float*)d_in[pi++];

  float* out = (float*)d_out;
  float* hs_out   = out;                         // [NL][BQ][C]
  float* refs_out = out + (long long)NLn*BQ*Cc;  // [NL+1][BQ][4]

  // workspace bump allocator (256B aligned)
  char* ws = (char*)d_ws; size_t off = 0;
  auto carve = [&](size_t bytes) -> void* {
    void* p = ws + off; off = (off + bytes + 255) & ~size_t(255); return p;
  };
  float*  value   = (float*)carve((size_t)Bc*LM*Cc*4);
  __bf16* mem_bf  = (__bf16*)carve((size_t)Bc*LM*Cc*2);
  float*  probs   = (float*)carve((size_t)BHn*NQ*KEYP32*4);
  float*  ffbuf   = (float*)carve((size_t)BQ*DFF*4);
  float*  qse     = (float*)carve((size_t)BQ*512*4);
  float*  qkbuf   = (float*)carve((size_t)BQ*512*4);
  float*  tgt     = (float*)carve((size_t)BQ*Cc*4);
  float*  h1      = (float*)carve((size_t)BQ*Cc*4);
  float*  qpos    = (float*)carve((size_t)BQ*Cc*4);
  float*  qin     = (float*)carve((size_t)BQ*Cc*4);
  float*  vbuf    = (float*)carve((size_t)BQ*Cc*4);
  float*  t2      = (float*)carve((size_t)BQ*Cc*4);
  float*  t2a     = (float*)carve((size_t)BQ*Cc*4);
  float*  tgt_n2  = (float*)carve((size_t)BQ*Cc*4);
  float*  tgt_n1  = (float*)carve((size_t)BQ*Cc*4);
  float*  sampled = (float*)carve((size_t)BQ*Cc*4);
  float*  hb1     = (float*)carve((size_t)BQ*Cc*4);
  float*  hb2     = (float*)carve((size_t)BQ*Cc*4);
  float*  so_out  = (float*)carve((size_t)BQ*Cc*4);
  float*  aw_out  = (float*)carve((size_t)BQ*128*4);
  float*  delta   = (float*)carve((size_t)BQ*4*4);
  float*  refcur  = (float*)carve((size_t)BQ*4*4);
  float*  qh      = (float*)carve((size_t)BHn*NQ*DH*4);
  float*  attnout = (float*)carve((size_t)BHn*NQ*DH*4);
  __bf16* khT     = (__bf16*)carve((size_t)BHn*KEYP32*DH*2);
  __bf16* vhT     = (__bf16*)carve((size_t)BHn*DH*KEYP32*2);
  __bf16* rph1T   = (__bf16*)carve((size_t)256*512*2);
  __bf16* rph2T   = (__bf16*)carve((size_t)256*256*2);
  __bf16* sainT   = (__bf16*)carve((size_t)768*256*2);
  __bf16* saoutT  = (__bf16*)carve((size_t)256*256*2);
  __bf16* soT     = (__bf16*)carve((size_t)256*256*2);
  __bf16* awT     = (__bf16*)carve((size_t)128*256*2);
  __bf16* vpT     = (__bf16*)carve((size_t)256*256*2);
  __bf16* opT     = (__bf16*)carve((size_t)256*256*2);
  __bf16* ff1T    = (__bf16*)carve((size_t)2048*256*2);
  __bf16* ff2T    = (__bf16*)carve((size_t)256*2048*2);
  __bf16* bb1T    = (__bf16*)carve((size_t)256*256*2);
  __bf16* bb2T    = (__bf16*)carve((size_t)256*256*2);
  __bf16* bb3T    = (__bf16*)carve((size_t)32*256*2);

  auto conv = [&](const float* W, __bf16* WT, int K, int N) {
    int Np = (N + 31) & ~31, Kp = (K + 31) & ~31;
    int total = Np * Kp;
    convert_wt<<<(total + 255)/256, 256, 0, stream>>>(W, WT, K, N, Kp, Np);
  };
  auto gemmF = [&](const float* A, const __bf16* WT, const float* bias, float* Co,
                   int M, int N, int Kp, int lda, int ldwt, int ldc,
                   long long sA, long long sW, long long sC, int batch,
                   float alpha, bool relu) {
    dim3 blk(32, 4), grd((M + 63)/64, (N + 31)/32, batch);
    if (relu)
      gemm_wmma<true , float><<<grd, blk, 0, stream>>>(A, WT, bias, Co, M, N, Kp,
                                                       lda, ldwt, ldc, sA, sW, sC, alpha);
    else
      gemm_wmma<false, float><<<grd, blk, 0, stream>>>(A, WT, bias, Co, M, N, Kp,
                                                       lda, ldwt, ldc, sA, sW, sC, alpha);
  };
  auto gemmB = [&](const __bf16* A, const __bf16* WT, const float* bias, float* Co,
                   int M, int N, int Kp, int lda, int ldwt, int ldc, float alpha) {
    dim3 blk(32, 4), grd((M + 63)/64, (N + 31)/32, 1);
    gemm_wmma<false, __bf16><<<grd, blk, 0, stream>>>(A, WT, bias, Co, M, N, Kp,
                                                      lda, ldwt, ldc, 0, 0, 0, alpha);
  };

  // once-per-call prep: shared weights, bf16 memory copy, initial state
  conv(rph_w1, rph1T, 512, 256);
  conv(rph_w2, rph2T, 256, 256);
  convert_bf<<<((Bc*LM*Cc) + 255)/256, 256, 0, stream>>>(memory, mem_bf, Bc*LM*Cc);
  hipMemcpyAsync(tgt, tgt_in, (size_t)BQ*Cc*4, hipMemcpyDeviceToDevice, stream);
  sigmoid0<<<(BQ*4 + 255)/256, 256, 0, stream>>>(rpu, refcur, refs_out, BQ*4);

  const float attn_scale = 0.1767766952966369f;   // 1/sqrt(DH)

  for (int i = 0; i < NLn; ++i) {
    // per-layer weights -> transposed bf16
    conv(sain_w + (size_t)i*256*768, sainT, 256, 768);
    conv(saout_w+ (size_t)i*256*256, saoutT,256, 256);
    conv(so_w  + (size_t)i*256*256, soT,   256, 256);
    conv(aw_w  + (size_t)i*256*128, awT,   256, 128);
    conv(vp_w  + (size_t)i*256*256, vpT,   256, 256);
    conv(op_w  + (size_t)i*256*256, opT,   256, 256);
    conv(ff1_w + (size_t)i*256*2048,ff1T,  256, 2048);
    conv(ff2_w + (size_t)i*2048*256,ff2T,  2048,256);
    conv(bb_w1 + (size_t)i*256*256, bb1T,  256, 256);
    conv(bb_w2 + (size_t)i*256*256, bb2T,  256, 256);
    conv(bb_w3 + (size_t)i*256*4,   bb3T,  256, 4);

    // query positional embedding (sine -> 2-layer MLP)
    sine_embed_k<<<BQ, 128, 0, stream>>>(refcur, vratio, qse);
    gemmF(qse, rph1T, rph_b1, h1,   BQ, 256, 512, 512, 512, 256, 0,0,0, 1, 1.f, true);
    gemmF(h1,  rph2T, rph_b2, qpos, BQ, 256, 256, 256, 256, 256, 0,0,0, 1, 1.f, false);

    // --- self attention ---
    ew_add<<<(BQ*Cc + 255)/256, 256, 0, stream>>>(tgt, qpos, qin, BQ*Cc);
    gemmF(qin, sainT,            sain_b + (size_t)i*768,       qkbuf, BQ, 512, 256, 256, 256, 512, 0,0,0, 1, 1.f, false);
    gemmF(tgt, sainT + 512*256,  sain_b + (size_t)i*768 + 512, vbuf,  BQ, 256, 256, 256, 256, 256, 0,0,0, 1, 1.f, false);
    prep_qh<<<(BHn*NQ*DH    + 255)/256, 256, 0, stream>>>(qkbuf, qh);
    prep_kh<<<(BHn*KEYP32*DH+ 255)/256, 256, 0, stream>>>(qkbuf, khT);
    prep_vh<<<(BHn*DH*KEYP32+ 255)/256, 256, 0, stream>>>(vbuf, vhT);
    // scores = (qh @ kh^T)/sqrt(DH) : batched over 16 (b,h)
    gemmF(qh, khT, nullptr, probs, NQ, NQ, DH, DH, DH, KEYP32,
          (long long)NQ*DH, (long long)KEYP32*DH, (long long)NQ*KEYP32, BHn, attn_scale, false);
    attn_softmax<<<BHn*NQ, KEYP32, 0, stream>>>(probs);
    // attn = probs @ vh
    gemmF(probs, vhT, nullptr, attnout, NQ, DH, KEYP32, KEYP32, KEYP32, DH,
          (long long)NQ*KEYP32, (long long)DH*KEYP32, (long long)NQ*DH, BHn, 1.f, false);
    merge_heads<<<(BHn*NQ*DH + 255)/256, 256, 0, stream>>>(attnout, t2a);
    gemmF(t2a, saoutT, saout_b + (size_t)i*256, t2, BQ, 256, 256, 256, 256, 256, 0,0,0, 1, 1.f, false);
    add_ln<<<BQ, 256, 0, stream>>>(tgt, t2, n2_g + (size_t)i*256, n2_b + (size_t)i*256, tgt_n2);

    // --- deformable cross attention ---
    ew_add<<<(BQ*Cc + 255)/256, 256, 0, stream>>>(tgt_n2, qpos, qin, BQ*Cc);
    gemmF(qin, soT, so_b + (size_t)i*256, so_out, BQ, 256, 256, 256, 256, 256, 0,0,0, 1, 1.f, false);
    gemmF(qin, awT, aw_b + (size_t)i*128, aw_out, BQ, 128, 256, 256, 256, 128, 0,0,0, 1, 1.f, false);
    aw_softmax<<<(BQ*Hh + 255)/256, 256, 0, stream>>>(aw_out);
    // value projection — the dominant GEMM (43520 x 256 x 256), bf16 A from L2
    gemmB(mem_bf, vpT, vp_b + (size_t)i*256, value, Bc*LM, 256, 256, 256, 256, 256, 1.f);
    ms_deform<<<BQ, dim3(32, 8), 0, stream>>>(value, so_out, aw_out, refcur, vratio, sampled);
    gemmF(sampled, opT, op_b + (size_t)i*256, t2, BQ, 256, 256, 256, 256, 256, 0,0,0, 1, 1.f, false);
    add_ln<<<BQ, 256, 0, stream>>>(tgt_n2, t2, n1_g + (size_t)i*256, n1_b + (size_t)i*256, tgt_n1);

    // --- FFN ---
    gemmF(tgt_n1, ff1T, ff1_b + (size_t)i*2048, ffbuf, BQ, 2048, 256, 256, 256, 2048, 0,0,0, 1, 1.f, true);
    gemmF(ffbuf, ff2T, ff2_b + (size_t)i*256, t2, BQ, 256, 2048, 2048, 2048, 256, 0,0,0, 1, 1.f, false);
    add_ln<<<BQ, 256, 0, stream>>>(tgt_n1, t2, n3_g + (size_t)i*256, n3_b + (size_t)i*256, tgt);

    // --- bbox head + ref update ---
    gemmF(tgt, bb1T, bb_b1 + (size_t)i*256, hb1, BQ, 256, 256, 256, 256, 256, 0,0,0, 1, 1.f, true);
    gemmF(hb1, bb2T, bb_b2 + (size_t)i*256, hb2, BQ, 256, 256, 256, 256, 256, 0,0,0, 1, 1.f, true);
    gemmF(hb2, bb3T, bb_b3 + (size_t)i*4,   delta, BQ, 4, 256, 256, 256, 4, 0,0,0, 1, 1.f, false);
    update_ref<<<(BQ*4 + 255)/256, 256, 0, stream>>>(delta, refcur,
                                                     refs_out + (size_t)(i+1)*BQ*4);
    // hs[i] = LayerNorm(out)
    add_ln<<<BQ, 256, 0, stream>>>(tgt, nullptr, dn_g, dn_b, hs_out + (size_t)i*BQ*Cc);
  }
}
